// DLSMNLayer_20590073217491
// MI455X (gfx1250) — matrix operand order repaired and checked by reading, not compiled, not run on hardware
//
#include <hip/hip_runtime.h>
#include <math.h>

// ---------------------------------------------------------------------------
// CDNA5 (gfx1250) implementation of the DLSMN layer.
// All dense math runs through v_wmma_f32_16x16x32_f16 (f16 in, f32 accum):
// the workload is ~320 GFLOP of GEMMs vs ~hundreds MB of traffic, so it is
// matrix-core bound on MI455X; f16 WMMA w/ fp32 accumulation is the fast path.
// GEMM staging is branch-free: clamped row addresses + unconditional 16B
// global loads + masked zeroing (requires K % 32 == 0, true for every call
// here), 16B LDS stores, and global_prefetch of the next K tile.
// Workspace peak use: ~52.8M floats (~212 MB).
// ---------------------------------------------------------------------------

typedef __attribute__((ext_vector_type(16))) _Float16 v16h;
typedef __attribute__((ext_vector_type(8)))  _Float16 v8h;
typedef __attribute__((ext_vector_type(8)))  float    v8f;
typedef __attribute__((ext_vector_type(4)))  float    v4f;

#define TM 64
#define TN 128
#define TK 32

struct GemmP {
  const float* A; long lda, sAo, sAi;
  const float* B; long ldb, sBo, sBi;
  float*       C; long ldc, sCo, sCi;
  const float* bias;                       // per-output-column bias or null
  const float* Add; long ldadd, sAddo, sAddi;  // optional matrix added to result
  int M, N, K, innerN;
  float alpha;
  int transB, act;                         // act: 0 none, 1 gelu(exact), 2 sigmoid
};

__device__ __forceinline__ float act_apply(float v, int act) {
  if (act == 1) return 0.5f * v * (1.0f + erff(v * 0.70710678118654752f));
  if (act == 2) return 1.0f / (1.0f + __expf(-v));
  return v;
}

// Block: 256 threads = 8 waves (wave32). Block tile 64x128, wave tile 32x32
// (2x2 WMMA 16x16 fragments). K staged through LDS in 32-wide f16 tiles.
// REQUIREMENT: p.K % 32 == 0 (no per-K guards; rows are clamped+masked).
__global__ __launch_bounds__(256) void gemm_wmma(GemmP p) {
  const int z  = blockIdx.z;
  const int zo = z / p.innerN, zi = z % p.innerN;
  const float* A = p.A + (long)zo * p.sAo + (long)zi * p.sAi;
  const float* B = p.B + (long)zo * p.sBo + (long)zi * p.sBi;
  float*       C = p.C + (long)zo * p.sCo + (long)zi * p.sCi;
  const float* Add = p.Add ? (p.Add + (long)zo * p.sAddo + (long)zi * p.sAddi) : nullptr;

  __shared__ _Float16 As[TM][TK + 8];   // row pitch 40 halves = 80B (16B aligned)
  __shared__ _Float16 Bs[TN][TK + 8];

  const int t = threadIdx.x, lane = t & 31, w = t >> 5;
  const int wm = w >> 2, wn = w & 3;           // waves: 2 (M) x 4 (N)
  const int ml = lane & 15, half = lane >> 4;  // WMMA lane decomposition
  const int bm = blockIdx.y * TM, bn = blockIdx.x * TN;

  // --- A staging thread mapping: 8 contiguous floats per thread ---
  const int ar = t >> 2;                 // tile row 0..63
  const int ac = (t & 3) * 8;            // tile col 0,8,16,24
  const int agm = bm + ar;
  const long arow = (long)(agm < p.M ? agm : (p.M - 1)) * p.lda;
  const float asel = (agm < p.M) ? 1.0f : 0.0f;

  // --- B staging thread mapping: 16 contiguous k per thread ---
  const int br = t >> 1;                 // tile row (n) 0..127
  const int bc = (t & 1) * 16;           // k offset 0 or 16
  const int bgn = bn + br;
  const int bgnc = (bgn < p.N) ? bgn : (p.N - 1);
  const float bsel = (bgn < p.N) ? 1.0f : 0.0f;

  v8f acc[2][2] = {};

  for (int k0 = 0; k0 < p.K; k0 += TK) {
    { // A tile: two unconditional 16B loads, masked, packed 16B LDS store
      const float* ap = A + arow + k0 + ac;
      v4f x0 = *(const v4f*)ap;
      v4f x1 = *(const v4f*)(ap + 4);
      if (k0 + TK < p.K) __builtin_prefetch(ap + TK, 0, 0);
      v8h hv;
      #pragma unroll
      for (int j = 0; j < 4; ++j) {
        hv[j]     = (_Float16)(x0[j] * asel);
        hv[j + 4] = (_Float16)(x1[j] * asel);
      }
      *(v8h*)(&As[ar][ac]) = hv;
    }
    if (p.transB) { // B tile rows are contiguous in k: four 16B loads
      const float* bp = B + (long)bgnc * p.ldb + k0 + bc;
      v4f y0 = *(const v4f*)bp;
      v4f y1 = *(const v4f*)(bp + 4);
      v4f y2 = *(const v4f*)(bp + 8);
      v4f y3 = *(const v4f*)(bp + 12);
      if (k0 + TK < p.K) __builtin_prefetch(bp + TK, 0, 0);
      v8h h0, h1;
      #pragma unroll
      for (int j = 0; j < 4; ++j) {
        h0[j]     = (_Float16)(y0[j] * bsel);
        h0[j + 4] = (_Float16)(y1[j] * bsel);
        h1[j]     = (_Float16)(y2[j] * bsel);
        h1[j + 4] = (_Float16)(y3[j] * bsel);
      }
      *(v8h*)(&Bs[br][bc])     = h0;
      *(v8h*)(&Bs[br][bc + 8]) = h1;
    } else {        // B is [K,N]: strided gather down the column, unconditional
      const float* bp = B + (long)(k0 + bc) * p.ldb + bgnc;
      float yv[16];
      #pragma unroll
      for (int j = 0; j < 16; ++j) yv[j] = bp[(long)j * p.ldb];
      v8h h0, h1;
      #pragma unroll
      for (int j = 0; j < 8; ++j) {
        h0[j] = (_Float16)(yv[j] * bsel);
        h1[j] = (_Float16)(yv[j + 8] * bsel);
      }
      *(v8h*)(&Bs[br][bc])     = h0;
      *(v8h*)(&Bs[br][bc + 8]) = h1;
    }
    __syncthreads();

    // Fragment loads per CDNA5 ISA 7.12.2 layouts (16-bit A 16x32; B 32x16).
    v16h af[2], bf[2];
    #pragma unroll
    for (int mt = 0; mt < 2; ++mt) {
      int row = wm * 32 + mt * 16 + ml;
      #pragma unroll
      for (int e = 0; e < 8; ++e) {
        af[mt][e]     = As[row][8 * half + e];        // K = 8*half + e
        af[mt][e + 8] = As[row][16 + 8 * half + e];   // K = 16 + 8*half + e
      }
    }
    #pragma unroll
    for (int nt = 0; nt < 2; ++nt) {
      int col = wn * 32 + nt * 16 + ml;
      #pragma unroll
      for (int e = 0; e < 8; ++e) {
        bf[nt][e]     = Bs[col][8 * half + e];        // K = 8*half + e
        bf[nt][e + 8] = Bs[col][16 + 8 * half + e];   // K = 16 + 8*half + e
      }
    }
    #pragma unroll
    for (int mt = 0; mt < 2; ++mt)
      #pragma unroll
      for (int nt = 0; nt < 2; ++nt)
        acc[mt][nt] = __builtin_amdgcn_wmma_f32_16x16x32_f16(
            false, af[mt], false, bf[nt], (short)0, acc[mt][nt], false, false);
    __syncthreads();
  }

  // Epilogue: 32-bit C/D layout -> lane(ml), row r + 8*half.
  #pragma unroll
  for (int mt = 0; mt < 2; ++mt) {
    #pragma unroll
    for (int nt = 0; nt < 2; ++nt) {
      int gn = bn + wn * 32 + nt * 16 + ml;
      if (gn >= p.N) continue;
      #pragma unroll
      for (int r = 0; r < 8; ++r) {
        int gm = bm + wm * 32 + mt * 16 + r + 8 * half;
        if (gm >= p.M) continue;
        float v = p.alpha * acc[mt][nt][r];
        if (p.bias) v += p.bias[gn];
        if (Add)    v += Add[(long)gm * p.ldadd + gn];
        C[(long)gm * p.ldc + gn] = act_apply(v, p.act);
      }
    }
  }
}

// Wave-per-row softmax (contiguous rows, stride == len).
__global__ __launch_bounds__(256) void softmax_rows(const float* in, float* out,
                                                    int rows, int len) {
  int row = blockIdx.x * 8 + (threadIdx.x >> 5);
  if (row >= rows) return;
  int lane = threadIdx.x & 31;
  const float* x = in + (long)row * len;
  float* y = out + (long)row * len;
  float mx = -3.4e38f;
  for (int i = lane; i < len; i += 32) mx = fmaxf(mx, x[i]);
  #pragma unroll
  for (int o = 16; o > 0; o >>= 1) mx = fmaxf(mx, __shfl_xor(mx, o, 32));
  float s = 0.0f;
  for (int i = lane; i < len; i += 32) s += __expf(x[i] - mx);
  #pragma unroll
  for (int o = 16; o > 0; o >>= 1) s += __shfl_xor(s, o, 32);
  float inv = 1.0f / s;
  for (int i = lane; i < len; i += 32) y[i] = __expf(x[i] - mx) * inv;
}

// Wave-per-row LayerNorm of (a + b), d fixed at 1024 (32 elems/lane).
__global__ __launch_bounds__(256) void ln_residual(const float* a, const float* b,
                                                   const float* g, const float* be,
                                                   float* out, int rows) {
  int row = blockIdx.x * 8 + (threadIdx.x >> 5);
  if (row >= rows) return;
  int lane = threadIdx.x & 31;
  const int d = 1024;
  float v[32];
  float s = 0.0f;
  #pragma unroll
  for (int j = 0; j < 32; ++j) {
    int i = lane + j * 32;
    v[j] = a[(long)row * d + i] + b[(long)row * d + i];
    s += v[j];
  }
  #pragma unroll
  for (int o = 16; o > 0; o >>= 1) s += __shfl_xor(s, o, 32);
  float mean = s * (1.0f / d);
  float vs = 0.0f;
  #pragma unroll
  for (int j = 0; j < 32; ++j) { float t2 = v[j] - mean; vs += t2 * t2; }
  #pragma unroll
  for (int o = 16; o > 0; o >>= 1) vs += __shfl_xor(vs, o, 32);
  float inv = rsqrtf(vs * (1.0f / d) + 1e-5f);
  #pragma unroll
  for (int j = 0; j < 32; ++j) {
    int i = lane + j * 32;
    out[(long)row * d + i] = (v[j] - mean) * inv * g[i] + be[i];
  }
}

// decb[n] = dec_b[n] + dec_w[n, 256:320] . layer_embed
__global__ void decb_kernel(const float* dec_w, const float* dec_b,
                            const float* le, float* outd) {
  int n = blockIdx.x * blockDim.x + threadIdx.x;
  if (n >= 1024) return;
  float s = dec_b[n];
  const float* wrow = dec_w + (long)n * 320 + 256;
  #pragma unroll 8
  for (int j = 0; j < 64; ++j) s += wrow[j] * le[j];
  outd[n] = s;
}

// xf = sigmoid(gpre)*x + (1-sigmoid(gpre))*context
__global__ void gate_fuse(const float* gpre, const float* x, const float* ctx,
                          float* xf, long n) {
  long i = (long)blockIdx.x * blockDim.x + threadIdx.x;
  if (i >= n) return;
  float g = 1.0f / (1.0f + __expf(-gpre[i]));
  xf[i] = g * x[i] + (1.0f - g) * ctx[i];
}

// logits = alpha[row]*learned + (1-alpha[row])*content  (row = i/64)
__global__ void logits_comb(const float* alpha, const float* learned,
                            const float* content, float* logits, int n) {
  int i = blockIdx.x * blockDim.x + threadIdx.x;
  if (i >= n) return;
  float a = alpha[i >> 6];
  logits[i] = a * learned[i] + (1.0f - a) * content[i];
}

// One wave per row of 64: slot = softmax(logits + gumbel(u)), soft = softmax(logits)
__global__ __launch_bounds__(256) void gumbel_softmax(const float* logits, const float* u,
                                                      float* slot, float* soft, int rows) {
  int row = blockIdx.x * 8 + (threadIdx.x >> 5);
  if (row >= rows) return;
  int lane = threadIdx.x & 31;
  float l0 = logits[row * 64 + lane], l1 = logits[row * 64 + 32 + lane];
  float u0 = u[row * 64 + lane],      u1 = u[row * 64 + 32 + lane];
  float g0 = -logf(-logf(u0 + 1e-8f) + 1e-8f);
  float g1 = -logf(-logf(u1 + 1e-8f) + 1e-8f);

  float a0 = l0 + g0, a1 = l1 + g1;
  float mx = fmaxf(a0, a1);
  #pragma unroll
  for (int o = 16; o > 0; o >>= 1) mx = fmaxf(mx, __shfl_xor(mx, o, 32));
  float e0 = __expf(a0 - mx), e1 = __expf(a1 - mx);
  float s = e0 + e1;
  #pragma unroll
  for (int o = 16; o > 0; o >>= 1) s += __shfl_xor(s, o, 32);
  slot[row * 64 + lane]      = e0 / s;
  slot[row * 64 + 32 + lane] = e1 / s;

  mx = fmaxf(l0, l1);
  #pragma unroll
  for (int o = 16; o > 0; o >>= 1) mx = fmaxf(mx, __shfl_xor(mx, o, 32));
  e0 = __expf(l0 - mx); e1 = __expf(l1 - mx);
  s = e0 + e1;
  #pragma unroll
  for (int o = 16; o > 0; o >>= 1) s += __shfl_xor(s, o, 32);
  soft[row * 64 + lane]      = e0 / s;
  soft[row * 64 + 32 + lane] = e1 / s;
}

static void launch_gemm(hipStream_t st,
    const float* A, long lda, long sAo, long sAi,
    const float* B, long ldb, long sBo, long sBi, int transB,
    float* C, long ldc, long sCo, long sCi,
    const float* bias, const float* Add, long ldadd, long sAddo, long sAddi,
    int M, int N, int K, int outerN, int innerN, float alpha, int act) {
  GemmP p{A, lda, sAo, sAi, B, ldb, sBo, sBi, C, ldc, sCo, sCi,
          bias, Add, ldadd, sAddo, sAddi, M, N, K, innerN, alpha, transB, act};
  dim3 grid((N + TN - 1) / TN, (M + TM - 1) / TM, outerN * innerN);
  gemm_wmma<<<grid, dim3(256), 0, st>>>(p);
}

extern "C" void kernel_launch(void* const* d_in, const int* in_sizes, int n_in,
                              void* d_out_v, int out_size, void* d_ws, size_t ws_size,
                              hipStream_t stream) {
  (void)in_sizes; (void)n_in; (void)out_size; (void)ws_size;

  const float* x            = (const float*)d_in[0];
  const float* cache        = (const float*)d_in[1];
  const float* slot_emb     = (const float*)d_in[2];
  const float* layer_ids    = (const float*)d_in[3];
  const float* gumbel_u     = (const float*)d_in[4];
  const float* layer_embed  = (const float*)d_in[5];
  const float* attn_wqkv    = (const float*)d_in[6];
  const float* attn_bqkv    = (const float*)d_in[7];
  const float* attn_wo      = (const float*)d_in[8];
  const float* attn_bo      = (const float*)d_in[9];
  const float* pat_wqkv     = (const float*)d_in[10];
  const float* pat_bqkv     = (const float*)d_in[11];
  const float* pat_wo       = (const float*)d_in[12];
  const float* pat_bo       = (const float*)d_in[13];
  const float* pattern_q    = (const float*)d_in[14];
  const float* ffn_w1       = (const float*)d_in[15];
  const float* ffn_b1       = (const float*)d_in[16];
  const float* ffn_w2       = (const float*)d_in[17];
  const float* ffn_b2       = (const float*)d_in[18];
  const float* n1_g         = (const float*)d_in[19];
  const float* n1_b         = (const float*)d_in[20];
  const float* n2_g         = (const float*)d_in[21];
  const float* n2_b         = (const float*)d_in[22];
  const float* gate_w       = (const float*)d_in[23];
  const float* gate_b       = (const float*)d_in[24];
  const float* sel_w        = (const float*)d_in[25];
  const float* sel_b        = (const float*)d_in[26];
  const float* sq_w         = (const float*)d_in[27];
  const float* sq_b         = (const float*)d_in[28];
  const float* sk_w         = (const float*)d_in[29];
  const float* sk_b         = (const float*)d_in[30];
  const float* an_w         = (const float*)d_in[31];
  const float* an_b         = (const float*)d_in[32];
  const float* dec_w        = (const float*)d_in[33];
  const float* dec_b        = (const float*)d_in[34];
  const float* cq_w         = (const float*)d_in[35];
  const float* cq_b         = (const float*)d_in[36];
  const float* ck_w         = (const float*)d_in[37];
  const float* ck_b         = (const float*)d_in[38];
  const float* cv_w         = (const float*)d_in[39];
  const float* cv_b         = (const float*)d_in[40];
  const float* fg_w         = (const float*)d_in[41];
  const float* fg_b         = (const float*)d_in[42];

  float* outp = (float*)d_out_v;
  float* ws   = (float*)d_ws;

  const int Sc = 1024, Dc = 1024;

  // workspace regions (floats), reused across phases
  float* R0 = ws;                 // 10,485,760
  float* R1 = ws + 10485760;      //  8,388,608
  float* R2 = ws + 18874368;      // 16,777,216
  float* R3 = ws + 35651584;      //  8,388,608
  float* R4 = ws + 44040192;      //  8,388,608
  float* SMp = ws + 52428800;     // small buffers

  float* q_rc  = R0;
  float* kk    = R0 + 2097152;
  float* vv    = R0 + 3145728;
  float* scb   = R0 + 4194304;
  float* ctx   = R0 + 8388608;
  float* decb  = SMp;
  float* kbias = SMp + 1024;
  float* vbias = SMp + 132096;
  float* qh0   = SMp + 263168;
  float* learned = SMp + 279552;
  float* sqp   = SMp + 287744;
  float* skk   = SMp + 320512;
  float* content = SMp + 336896;
  float* logits  = SMp + 345088;
  float* xf      = R1;
  float* contextb = R3;
  float* gpre    = R4;
  float* qkvb    = R0;    // per-batch qkv [1024,3072]
  float* sbuf    = R2;    // [16,1024,1024]
  float* attnout = R3;
  float* proj    = R4;
  float* y1      = R1;
  float* hbuf    = R0;    // FFN hidden chunk [2048,4096]
  float* ffnout  = R4;
  float* kvbuf   = R2;    // pattern k|v [8192,2048]
  float* psc     = R0;    // pattern scores [8,16,16,1024]
  float* ppre    = R1;    // pattern attn out [8,16,1024]

  float* out_y    = outp;
  float* out_pat  = outp + 8388608;
  float* out_sc   = outp + 8519680;
  float* out_slot = outp + 8519808;
  float* out_soft = outp + 8528000;
  float* out_al   = outp + 8536192;

  // ---- read_cache ----
  decb_kernel<<<dim3(4), dim3(256), 0, stream>>>(dec_w, dec_b, layer_embed, decb);
  // kbias/vbias = layer_ids @ W[:,256:]^T + b   (fold concat into bias matrix)
  launch_gemm(stream, layer_ids,64,0,0, ck_w+256,320,0,0,1, kbias,256,0,0,
              ck_b, nullptr,0,0,0, 512,256,64, 1,1, 1.0f, 0);
  launch_gemm(stream, layer_ids,64,0,0, cv_w+256,320,0,0,1, vbias,256,0,0,
              cv_b, nullptr,0,0,0, 512,256,64, 1,1, 1.0f, 0);
  // q = x @ cq_w^T + cq_b
  launch_gemm(stream, x,1024,0,0, cq_w,1024,0,0,1, q_rc,256,0,0,
              cq_b, nullptr,0,0,0, 8192,256,1024, 1,1, 1.0f, 0);
  // kk/vv = cache @ W[:,:256]^T + kbias/vbias  (batched over b)
  launch_gemm(stream, cache,256,131072,0, ck_w,320,0,0,1, kk,256,131072,0,
              nullptr, kbias,256,0,0, 512,256,256, 8,1, 1.0f, 0);
  launch_gemm(stream, cache,256,131072,0, cv_w,320,0,0,1, vv,256,131072,0,
              nullptr, vbias,256,0,0, 512,256,256, 8,1, 1.0f, 0);
  // sc = q @ kk^T / sqrt(256)
  launch_gemm(stream, q_rc,256,262144,0, kk,256,131072,0,1, scb,512,524288,0,
              nullptr, nullptr,0,0,0, 1024,512,256, 8,1, 0.0625f, 0);
  softmax_rows<<<dim3(1024), 256, 0, stream>>>(scb, scb, 8192, 512);
  // ctx = attn @ vv
  launch_gemm(stream, scb,512,524288,0, vv,256,131072,0,0, ctx,256,262144,0,
              nullptr, nullptr,0,0,0, 1024,256,512, 8,1, 1.0f, 0);
  // context = ctx @ dec_w[:,:256]^T + decb
  launch_gemm(stream, ctx,256,0,0, dec_w,320,0,0,1, contextb,1024,0,0,
              decb, nullptr,0,0,0, 8192,1024,256, 1,1, 1.0f, 0);

  // ---- gated fusion ----
  launch_gemm(stream, x,1024,0,0, fg_w,2048,0,0,1, gpre,1024,0,0,
              fg_b, nullptr,0,0,0, 8192,1024,1024, 1,1, 1.0f, 0);
  launch_gemm(stream, contextb,1024,0,0, fg_w+1024,2048,0,0,1, gpre,1024,0,0,
              nullptr, gpre,1024,0,0, 8192,1024,1024, 1,1, 1.0f, 0);
  gate_fuse<<<dim3(32768), 256, 0, stream>>>(gpre, x, contextb, xf, (long)8192*1024);

  // ---- self attention (per batch, scores materialized per b) ----
  for (int b = 0; b < 8; ++b) {
    launch_gemm(stream, xf + (long)b*Sc*Dc,1024,0,0, attn_wqkv,1024,0,0,1,
                qkvb,3072,0,0, attn_bqkv, nullptr,0,0,0, 1024,3072,1024, 1,1, 1.0f, 0);
    // scores[h] = q_h @ k_h^T / 8
    launch_gemm(stream, qkvb,3072,0,64, qkvb+1024,3072,0,64,1,
                sbuf,1024,0,1048576, nullptr, nullptr,0,0,0,
                1024,1024,64, 1,16, 0.125f, 0);
    softmax_rows<<<dim3(2048), 256, 0, stream>>>(sbuf, sbuf, 16384, 1024);
    // out[h] = P @ v_h
    launch_gemm(stream, sbuf,1024,0,1048576, qkvb+2048,3072,0,64,0,
                attnout + (long)b*Sc*Dc,1024,0,64, nullptr, nullptr,0,0,0,
                1024,64,1024, 1,16, 1.0f, 0);
  }
  launch_gemm(stream, attnout,1024,0,0, attn_wo,1024,0,0,1, proj,1024,0,0,
              attn_bo, nullptr,0,0,0, 8192,1024,1024, 1,1, 1.0f, 0);
  ln_residual<<<dim3(1024), 256, 0, stream>>>(xf, proj, n1_g, n1_b, y1, 8192);

  // ---- FFN (row-chunked hidden buffer) ----
  for (int c = 0; c < 4; ++c) {
    const float* yc = y1 + (long)c*2048*1024;
    launch_gemm(stream, yc,1024,0,0, ffn_w1,1024,0,0,1, hbuf,4096,0,0,
                ffn_b1, nullptr,0,0,0, 2048,4096,1024, 1,1, 1.0f, 1);
    launch_gemm(stream, hbuf,4096,0,0, ffn_w2,4096,0,0,1,
                ffnout + (long)c*2048*1024,1024,0,0,
                ffn_b2, nullptr,0,0,0, 2048,1024,4096, 1,1, 1.0f, 0);
  }
  ln_residual<<<dim3(1024), 256, 0, stream>>>(y1, ffnout, n2_g, n2_b, out_y, 8192);

  // ---- pattern pooling MHA ----
  // q proj of pattern queries (identical across batch -> compute once)
  launch_gemm(stream, pattern_q,1024,0,0, pat_wqkv,1024,0,0,1, qh0,1024,0,0,
              pat_bqkv, nullptr,0,0,0, 16,1024,1024, 1,1, 1.0f, 0);
  // k|v = y @ pat_wqkv[1024:3072]^T + b
  launch_gemm(stream, out_y,1024,0,0, pat_wqkv+1048576,1024,0,0,1, kvbuf,2048,0,0,
              pat_bqkv+1024, nullptr,0,0,0, 8192,2048,1024, 1,1, 1.0f, 0);
  // scores[b,h] = q_h @ k_{b,h}^T / 8   (outer=b, inner=h)
  launch_gemm(stream, qh0,1024,0,64, kvbuf,2048,2097152,64,1,
              psc,1024,262144,16384, nullptr, nullptr,0,0,0,
              16,1024,64, 8,16, 0.125f, 0);
  softmax_rows<<<dim3(256), 256, 0, stream>>>(psc, psc, 2048, 1024);
  launch_gemm(stream, psc,1024,262144,16384, kvbuf+1024,2048,2097152,64,0,
              ppre,1024,16384,64, nullptr, nullptr,0,0,0,
              16,64,1024, 8,16, 1.0f, 0);
  launch_gemm(stream, ppre,1024,0,0, pat_wo,1024,0,0,1, out_pat,1024,0,0,
              pat_bo, nullptr,0,0,0, 128,1024,1024, 1,1, 1.0f, 0);

  // ---- selection head ----
  launch_gemm(stream, out_pat,1024,0,0, gate_w,1024,0,0,1, out_sc,1,0,0,
              gate_b, nullptr,0,0,0, 128,1,1024, 1,1, 1.0f, 2);
  launch_gemm(stream, out_pat,1024,0,0, sel_w,1024,0,0,1, learned,64,0,0,
              sel_b, nullptr,0,0,0, 128,64,1024, 1,1, 1.0f, 0);
  launch_gemm(stream, out_pat,1024,0,0, sq_w,1024,0,0,1, sqp,256,0,0,
              sq_b, nullptr,0,0,0, 128,256,1024, 1,1, 1.0f, 0);
  launch_gemm(stream, slot_emb,256,0,0, sk_w,256,0,0,1, skk,256,0,0,
              sk_b, nullptr,0,0,0, 64,256,256, 1,1, 1.0f, 0);
  launch_gemm(stream, sqp,256,0,0, skk,256,0,0,1, content,64,0,0,
              nullptr, nullptr,0,0,0, 128,64,256, 1,1, 0.0625f, 0);
  launch_gemm(stream, out_pat,1024,0,0, an_w,1024,0,0,1, out_al,1,0,0,
              an_b, nullptr,0,0,0, 128,1,1024, 1,1, 1.0f, 2);
  logits_comb<<<dim3(32), 256, 0, stream>>>(out_al, learned, content, logits, 8192);
  gumbel_softmax<<<dim3(16), 256, 0, stream>>>(logits, gumbel_u, out_slot, out_soft, 128);
}
